// SatelliteGCN_63668595196286
// MI455X (gfx1250) — matrix operand order repaired and checked by reading, not compiled, not run on hardware
//
#include <hip/hip_runtime.h>
#include <hip/hip_bf16.h>
#include <stdint.h>

#define NN 100000
#define NE 1600000
#define DIM 128

typedef __attribute__((ext_vector_type(16))) __bf16 v16bf;
typedef __attribute__((ext_vector_type(8)))  __bf16 v8bf;
typedef __attribute__((ext_vector_type(8)))  float  v8f;

__device__ __forceinline__ __bf16 f2bf(float f) {
  uint32_t u = __builtin_bit_cast(uint32_t, f);
  uint32_t r = u + 0x7FFFu + ((u >> 16) & 1u);           // RNE
  uint16_t h = (uint16_t)(r >> 16);
  return __builtin_bit_cast(__bf16, h);
}
__device__ __forceinline__ float bf2f(__bf16 b) {
  uint32_t u = ((uint32_t)__builtin_bit_cast(uint16_t, b)) << 16;
  return __builtin_bit_cast(float, u);
}
__device__ __forceinline__ float bfbits2f(uint32_t hbits) {
  return __builtin_bit_cast(float, hbits << 16);
}

// ---------------- degree / normalization ----------------
__global__ void k_deg_init(unsigned* deg) {
  int i = blockIdx.x * blockDim.x + threadIdx.x;
  if (i < NN) deg[i] = 1u;                                // self-loop
}
__global__ void k_deg_edges(const long long* __restrict__ ei, unsigned* deg) {
  int e = blockIdx.x * blockDim.x + threadIdx.x;
  if (e < NE) atomicAdd(&deg[(int)ei[NE + e]], 1u);       // count on col
}
__global__ void k_dinv(const unsigned* __restrict__ deg, float* __restrict__ dinv) {
  int i = blockIdx.x * blockDim.x + threadIdx.x;
  if (i < NN) dinv[i] = rsqrtf((float)deg[i]);
}

// ---------------- GEMM: hs = dinv[row] * (A @ W), bf16 out ----------------
// block = 256 threads = 8 waves; 64 rows per block; wave w -> cols [16w,16w+16)
template <bool BF16IN>
__global__ void k_gemm_hs(const float* __restrict__ Af32,
                          const __bf16* __restrict__ Abf,
                          const float* __restrict__ W,      // [k][n] row-major
                          const float* __restrict__ dinv,
                          __bf16* __restrict__ hs) {
  __shared__ __bf16 Wt[DIM * DIM];   // transposed: [n][k], 32 KB
  __shared__ __bf16 Xs[64 * DIM];    // 16 KB: input tile, then reused as out tile
  __shared__ float  dls[64];
  const int tid = threadIdx.x;
  const int blockRow = blockIdx.x * 64;
  const bool fullBlock = (blockRow + 64 <= NN);

  #pragma unroll 4
  for (int i = 0; i < 64; ++i) {                 // 16384 W elements -> bf16^T
    int idx = i * 256 + tid;
    int k = idx >> 7, n = idx & 127;
    Wt[n * DIM + k] = f2bf(W[idx]);
  }
  if (BF16IN && fullBlock) {
    // CDNA5 async copy: global -> LDS without VGPR round-trip (ASYNCcnt).
    const __bf16* src = Abf + (size_t)blockRow * DIM;
    #pragma unroll
    for (int i = 0; i < 4; ++i) {
      int chunk = i * 256 + tid;                 // 1024 x 16B chunks = 16 KB
      unsigned lds_off = (unsigned)(uintptr_t)(&Xs[chunk * 8]);
      unsigned goff = (unsigned)chunk * 16u;
      asm volatile("global_load_async_to_lds_b128 %0, %1, %2"
                   :: "v"(lds_off), "v"(goff), "s"(src) : "memory");
    }
    asm volatile("s_wait_asynccnt 0x0" ::: "memory");
  } else {
    #pragma unroll 4
    for (int i = 0; i < 32; ++i) {               // 8192 X elements
      int idx = i * 256 + tid;
      int r = idx >> 7, k = idx & 127;
      int gr = blockRow + r;
      float v = 0.f;
      if (gr < NN) v = BF16IN ? bf2f(Abf[(size_t)gr * DIM + k])
                              : Af32[(size_t)gr * DIM + k];
      Xs[idx] = f2bf(v);
    }
  }
  if (tid < 64) {
    int gr = blockRow + tid;
    dls[tid] = (gr < NN) ? dinv[gr] : 0.f;
  }
  __syncthreads();

  const int wave = tid >> 5;
  const int lane = tid & 31;
  const int lo16 = lane & 15;
  const int hi   = lane >> 4;            // 0/1: lane-half selects K-slice
  const int n    = wave * 16 + lo16;     // output column for B/C/D

  // B fragments (same k order as A): col = lane%16, K-half by lane/16, 16 contiguous k
  v16bf bfrag[4];
  #pragma unroll
  for (int ks = 0; ks < 4; ++ks) {
    int kb = ks * 32 + hi * 16;
    bfrag[ks] = *reinterpret_cast<const v16bf*>(&Wt[n * DIM + kb]);
  }

  v8f accs[4];
  #pragma unroll
  for (int rt = 0; rt < 4; ++rt) {
    v8f acc = {0.f, 0.f, 0.f, 0.f, 0.f, 0.f, 0.f, 0.f};
    const int arow = rt * 16 + lo16;     // A: row = lane%16
    #pragma unroll
    for (int ks = 0; ks < 4; ++ks) {
      int kb = ks * 32 + hi * 16;        // A: K-half by lane-half, 16 contiguous k
      v16bf a = *reinterpret_cast<const v16bf*>(&Xs[arow * DIM + kb]);
      acc = __builtin_amdgcn_wmma_f32_16x16x32_bf16(
          false, a, false, bfrag[ks], (short)0, acc, false, false);
    }
    accs[rt] = acc;
  }

  __syncthreads();                       // all waves done reading Xs
  // Stage D tiles to LDS (scale by dinv, convert bf16), then coalesced stores.
  // C/D layout: VGPR vi, lanes 0-15 -> M=vi ; lanes 16-31 -> M=vi+8
  #pragma unroll
  for (int rt = 0; rt < 4; ++rt) {
    #pragma unroll
    for (int vi = 0; vi < 8; ++vi) {
      int ml = rt * 16 + hi * 8 + vi;
      Xs[ml * DIM + n] = f2bf(dls[ml] * accs[rt][vi]);
    }
  }
  __syncthreads();

  const float4* Ov = reinterpret_cast<const float4*>(Xs);
  float4* dst = reinterpret_cast<float4*>(hs + (size_t)blockRow * DIM);
  if (fullBlock) {
    #pragma unroll
    for (int i = 0; i < 4; ++i) {
      int chunk = i * 256 + tid;
      dst[chunk] = Ov[chunk];
    }
  } else {
    #pragma unroll
    for (int i = 0; i < 4; ++i) {
      int chunk = i * 256 + tid;
      if (blockRow + (chunk >> 4) < NN) dst[chunk] = Ov[chunk];
    }
  }
}

// ---------------- self-loop: acc = (float)hs ----------------
__global__ void k_selfloop(const __bf16* __restrict__ hs, float* __restrict__ acc) {
  int g = blockIdx.x * blockDim.x + threadIdx.x;     // NN*DIM/4 threads
  if (g >= NN * DIM / 4) return;
  uint2 p = reinterpret_cast<const uint2*>(hs)[g];
  float4 o = { bfbits2f(p.x & 0xffffu), bfbits2f(p.x >> 16),
               bfbits2f(p.y & 0xffffu), bfbits2f(p.y >> 16) };
  reinterpret_cast<float4*>(acc)[g] = o;
}

// ---------------- edge scatter: acc[c] += hs[r] (L2-resident atomics) ------
__global__ void k_scatter(const long long* __restrict__ ei,
                          const __bf16* __restrict__ hs,
                          float* __restrict__ acc) {
  int g = blockIdx.x * blockDim.x + threadIdx.x;
  int e = g >> 5, l = g & 31;                        // wave per edge, 4 dims/lane
  if (e >= NE) return;
  int r = (int)ei[e];
  int c = (int)ei[NE + e];
  uint2 p = reinterpret_cast<const uint2*>(hs + (size_t)r * DIM)[l];
  float* dst = acc + (size_t)c * DIM + l * 4;
  unsafeAtomicAdd(dst + 0, bfbits2f(p.x & 0xffffu));
  unsafeAtomicAdd(dst + 1, bfbits2f(p.x >> 16));
  unsafeAtomicAdd(dst + 2, bfbits2f(p.y & 0xffffu));
  unsafeAtomicAdd(dst + 3, bfbits2f(p.y >> 16));
}

// ---------------- epilogue: h = relu(dinv*acc + b) -> bf16 ----------------
__global__ void k_bias_relu(const float* __restrict__ acc,
                            const float* __restrict__ dinv,
                            const float* __restrict__ bias,
                            __bf16* __restrict__ h) {
  int g = blockIdx.x * blockDim.x + threadIdx.x;     // NN*DIM/4 threads
  if (g >= NN * DIM / 4) return;
  int i  = g >> 5;                                   // node
  int d  = (g & 31) * 4;                             // feature base
  float di = dinv[i];
  float4 v = reinterpret_cast<const float4*>(acc)[g];
  float r0 = fmaxf(di * v.x + bias[d + 0], 0.f);
  float r1 = fmaxf(di * v.y + bias[d + 1], 0.f);
  float r2 = fmaxf(di * v.z + bias[d + 2], 0.f);
  float r3 = fmaxf(di * v.w + bias[d + 3], 0.f);
  uint2 o;
  o.x = (uint32_t)__builtin_bit_cast(uint16_t, f2bf(r0)) |
        ((uint32_t)__builtin_bit_cast(uint16_t, f2bf(r1)) << 16);
  o.y = (uint32_t)__builtin_bit_cast(uint16_t, f2bf(r2)) |
        ((uint32_t)__builtin_bit_cast(uint16_t, f2bf(r3)) << 16);
  reinterpret_cast<uint2*>(h)[g] = o;
}

// ---------------- fused layer-2 epilogue + final 128->1 dot ----------------
__global__ void k_final(const float* __restrict__ acc,
                        const float* __restrict__ dinv,
                        const float* __restrict__ b2,
                        const float* __restrict__ Wo,
                        const float* __restrict__ bo,
                        float* __restrict__ out) {
  int g = blockIdx.x * blockDim.x + threadIdx.x;     // NN*32 threads
  int i = g >> 5, l = g & 31;
  if (i >= NN) return;
  float di = dinv[i];
  float4 v = reinterpret_cast<const float4*>(acc + (size_t)i * DIM)[l];
  int d = l * 4;
  float s = fmaxf(di * v.x + b2[d + 0], 0.f) * Wo[d + 0]
          + fmaxf(di * v.y + b2[d + 1], 0.f) * Wo[d + 1]
          + fmaxf(di * v.z + b2[d + 2], 0.f) * Wo[d + 2]
          + fmaxf(di * v.w + b2[d + 3], 0.f) * Wo[d + 3];
  #pragma unroll
  for (int off = 16; off > 0; off >>= 1) s += __shfl_xor(s, off, 32);
  if (l == 0) out[i] = s + bo[0];
}

extern "C" void kernel_launch(void* const* d_in, const int* in_sizes, int n_in,
                              void* d_out, int out_size, void* d_ws, size_t ws_size,
                              hipStream_t stream) {
  (void)in_sizes; (void)n_in; (void)out_size; (void)ws_size;
  const float*     x  = (const float*)d_in[0];
  const long long* ei = (const long long*)d_in[1];
  const float*     W1 = (const float*)d_in[2];
  const float*     b1 = (const float*)d_in[3];
  const float*     W2 = (const float*)d_in[4];
  const float*     b2 = (const float*)d_in[5];
  const float*     Wo = (const float*)d_in[6];
  const float*     bo = (const float*)d_in[7];
  float* out = (float*)d_out;

  char* ws = (char*)d_ws;
  size_t off = 0;
  auto alloc = [&](size_t bytes) {
    size_t p = off;
    off = (off + bytes + 255) & ~(size_t)255;
    return p;
  };
  float*    dinv = (float*)   (ws + alloc((size_t)NN * 4));
  unsigned* deg  = (unsigned*)(ws + alloc((size_t)NN * 4));
  __bf16*   bufA = (__bf16*)  (ws + alloc((size_t)NN * DIM * 2)); // hs (scaled)
  __bf16*   bufC = (__bf16*)  (ws + alloc((size_t)NN * DIM * 2)); // activations
  float*    bufB = (float*)   (ws + alloc((size_t)NN * DIM * 4)); // fp32 accum

  const int T = 256;
  const int gN    = (NN + T - 1) / T;
  const int gE    = (NE + T - 1) / T;
  const int gElem = (NN * DIM / 4 + T - 1) / T;   // 12500
  const int gEdgW = (NE * 32 + T - 1) / T;        // 200000
  const int gNodW = (NN * 32 + T - 1) / T;        // 12500
  const int gGemm = (NN + 63) / 64;               // 1563

  // normalization
  k_deg_init <<<gN, T, 0, stream>>>(deg);
  k_deg_edges<<<gE, T, 0, stream>>>(ei, deg);
  k_dinv     <<<gN, T, 0, stream>>>(deg, dinv);

  // layer 1
  k_gemm_hs<false><<<gGemm, T, 0, stream>>>(x, nullptr, W1, dinv, bufA);
  k_selfloop      <<<gElem, T, 0, stream>>>(bufA, bufB);
  k_scatter       <<<gEdgW, T, 0, stream>>>(ei, bufA, bufB);
  k_bias_relu     <<<gElem, T, 0, stream>>>(bufB, dinv, b1, bufC);

  // layer 2
  k_gemm_hs<true> <<<gGemm, T, 0, stream>>>(nullptr, bufC, W2, dinv, bufA);
  k_selfloop      <<<gElem, T, 0, stream>>>(bufA, bufB);
  k_scatter       <<<gEdgW, T, 0, stream>>>(ei, bufA, bufB);

  // fused epilogue + output projection
  k_final<<<gNodW, T, 0, stream>>>(bufB, dinv, b2, Wo, bo, out);
}
